// FastGeoCachedAttention_71545565216805
// MI455X (gfx1250) — matrix operand order repaired and checked by reading, not compiled, not run on hardware
//
#include <hip/hip_runtime.h>
#include <math.h>

#define EMBED   512
#define NHEADS  8
#define HD      64
#define NBINS   512
#define MAXBIN  256
#define RSCALE  0.125f   // 1/sqrt(64)

typedef __attribute__((ext_vector_type(16))) __bf16        bf16x16;
typedef __attribute__((ext_vector_type(8)))  float         f32x8;
typedef __attribute__((ext_vector_type(4)))  unsigned int  u32x4;

#define WMMA_BF16(a, b, c) \
  __builtin_amdgcn_wmma_f32_16x16x32_bf16(false, (a), false, (b), (short)0, (c), false, false)

static __device__ inline f32x8 zero8() {
  f32x8 z = {0.f, 0.f, 0.f, 0.f, 0.f, 0.f, 0.f, 0.f};
  return z;
}

// 16 contiguous bf16 (B-matrix fragment: lane holds 16 K-values for its column)
static __device__ inline bf16x16 ld_contig16(const __bf16* p) {
  union { u32x4 q[2]; bf16x16 v; } u;
  u.q[0] = *(const u32x4*)(p);
  u.q[1] = *(const u32x4*)(p + 8);
  return u.v;
}

// A-matrix fragment (16-bit A 16x32): 16 contiguous k at p (elems 0-7 at p, 8-15 at p+8)
static __device__ inline bf16x16 ld_afrag(const __bf16* p) {
  union { u32x4 q[2]; bf16x16 v; } u;
  u.q[0] = *(const u32x4*)(p);
  u.q[1] = *(const u32x4*)(p + 8);
  return u.v;
}

// ---------------------------------------------------------------- conversion
__global__ void cvt_f32_bf16(const float* __restrict__ in, __bf16* __restrict__ out, int n) {
  int i = blockIdx.x * blockDim.x + threadIdx.x;
  int stride = gridDim.x * blockDim.x;
  for (; i < n; i += stride) out[i] = (__bf16)in[i];
}

// ---------------------------------------------------------------- GEMM: C = A @ W^T + bias
// A: [M,K] bf16 row-major, W: [Nn,K] bf16 row-major (so B = W^T), bias fp32 [Nn]
// split=1 -> C written as [H][M][64] fp32 (head-split); split=0 -> [M,Nn] fp32
__global__ __launch_bounds__(256)
void gemm_wmma_bf16(const __bf16* __restrict__ A, const __bf16* __restrict__ W,
                    const float* __restrict__ bias, float* __restrict__ C,
                    int M, int Nn, int K, int split) {
  int l      = threadIdx.x & 31;
  int wid    = threadIdx.x >> 5;
  int ntile  = Nn >> 4;
  int w      = blockIdx.x * 8 + wid;
  int tn     = (w % ntile) << 4;
  int tm     = (w / ntile) << 4;
  if (tm >= M) return;

  int lane16 = l & 15;
  int half   = l >> 4;
  const __bf16* arow = A + (size_t)(tm + lane16) * K + half * 16;
  const __bf16* wrow = W + (size_t)(tn + lane16) * K + half * 16;

  f32x8 acc = zero8();
  for (int k0 = 0; k0 < K; k0 += 32) {
    __builtin_prefetch(arow + k0 + 256, 0, 0);   // global_prefetch_b8
    bf16x16 a = ld_afrag(arow + k0);
    bf16x16 b = ld_contig16(wrow + k0);
    acc = WMMA_BF16(a, b, acc);
  }

  int col = tn + lane16;
  float bv = bias[col];
#pragma unroll
  for (int r = 0; r < 8; ++r) {
    int row = tm + (half << 3) + r;
    float val = acc[r] + bv;
    if (split) {
      int h = col >> 6, j = col & 63;
      C[((size_t)h * M + row) * HD + j] = val;
    } else {
      C[(size_t)row * Nn + col] = val;
    }
  }
}

// ---------------------------------------------------------------- L2-normalize (wave per row)
__global__ __launch_bounds__(256)
void norm_kernel(const float* __restrict__ X, __bf16* __restrict__ Xn) {
  int wid = threadIdx.x >> 5, l = threadIdx.x & 31;
  int row = blockIdx.x * 8 + wid;
  const float* xp = X + (size_t)row * HD;
  float x0 = xp[2 * l], x1 = xp[2 * l + 1];
  float ss = x0 * x0 + x1 * x1;
#pragma unroll
  for (int off = 16; off; off >>= 1) ss += __shfl_xor(ss, off, 32);
  float inv = 1.0f / fmaxf(sqrtf(ss), 1e-12f);
  Xn[(size_t)row * HD + 2 * l]     = (__bf16)(x0 * inv);
  Xn[(size_t)row * HD + 2 * l + 1] = (__bf16)(x1 * inv);
}

// ---------------------------------------------------------------- WMMA centroid argmax
// Each wave: 16 rows x 512 centroids = 32 tiles x (2 WMMA, K=64); running argmax in C-layout.
__global__ __launch_bounds__(256)
void bin_wmma_kernel(const __bf16* __restrict__ Xn, const __bf16* __restrict__ cenb,
                     int* __restrict__ bins) {
  int l = threadIdx.x & 31, wid = threadIdx.x >> 5;
  int r0 = (blockIdx.x * 8 + wid) * 16;
  int lane16 = l & 15, half = l >> 4;

  const __bf16* xr = Xn + (size_t)(r0 + lane16) * HD + half * 16;
  bf16x16 a0 = ld_afrag(xr);
  bf16x16 a1 = ld_afrag(xr + 32);

  float best[8]; int bidx[8];
#pragma unroll
  for (int r = 0; r < 8; ++r) { best[r] = -1e30f; bidx[r] = 0; }

  for (int t = 0; t < NBINS / 16; ++t) {
    const __bf16* cp = cenb + (size_t)(t * 16 + lane16) * HD + half * 16;
    f32x8 s = zero8();
    s = WMMA_BF16(a0, ld_contig16(cp), s);
    s = WMMA_BF16(a1, ld_contig16(cp + 32), s);
    int c = t * 16 + lane16;
#pragma unroll
    for (int r = 0; r < 8; ++r)
      if (s[r] > best[r]) { best[r] = s[r]; bidx[r] = c; }  // strict >: first max kept
  }
#pragma unroll
  for (int r = 0; r < 8; ++r) {
    float b = best[r]; int bi = bidx[r];
#pragma unroll
    for (int off = 8; off; off >>= 1) {
      float ob = __shfl_xor(b, off, 16);
      int obi  = __shfl_xor(bi, off, 16);
      if (ob > b || (ob == b && obi < bi)) { b = ob; bi = obi; }
    }
    if (lane16 == 0) bins[r0 + (half << 3) + r] = bi;
  }
}

// ---------------------------------------------------------------- V transpose [H][N][HD]f32 -> [H][HD][N]bf16
__global__ __launch_bounds__(256)
void transpose_v_kernel(const float* __restrict__ Vf, __bf16* __restrict__ vt, int N) {
  __shared__ float tile[16][HD + 1];
  int h = blockIdx.y;
  int n0 = blockIdx.x * 16;
  for (int i = threadIdx.x; i < 16 * HD; i += 256) {
    int n = i >> 6, j = i & 63;
    tile[n][j] = Vf[((size_t)h * N + n0 + n) * HD + j];
  }
  __syncthreads();
  for (int i = threadIdx.x; i < 16 * HD; i += 256) {
    int j = i >> 4, n = i & 15;
    vt[((size_t)h * HD + j) * N + n0 + n] = (__bf16)tile[n][j];
  }
}

// ---------------------------------------------------------------- any-neighbor via bin occupancy bitmap
__global__ __launch_bounds__(256)
void neigh_any_kernel(const int* __restrict__ qb, const int* __restrict__ kb,
                      unsigned* __restrict__ any, int N) {
  __shared__ unsigned occ[NBINS / 32];
  int h = blockIdx.x;
  if (threadIdx.x < NBINS / 32) occ[threadIdx.x] = 0;
  __syncthreads();
  for (int k = threadIdx.x; k < N; k += 256) {
    int b = kb[(size_t)h * N + k];
    atomicOr(&occ[b >> 5], 1u << (b & 31));
  }
  __syncthreads();
  for (int q = threadIdx.x; q < N; q += 256) {
    int qv  = qb[(size_t)h * N + q];
    int bm1 = (qv - 1) & (NBINS - 1);
    int bp1 = (qv + 1) & (NBINS - 1);
    unsigned a = ((occ[qv >> 5] >> (qv & 31)) |
                  (occ[bm1 >> 5] >> (bm1 & 31)) |
                  (occ[bp1 >> 5] >> (bp1 & 31))) & 1u;
    any[(size_t)h * N + q] = a;
  }
}

// ---------------------------------------------------------------- flash attention + bin mask
// One wave = 16 q-rows of one head; 8 waves/block share the kb[] staging.
__global__ __launch_bounds__(256)
void attn_wmma_kernel(const __bf16* __restrict__ qn, const __bf16* __restrict__ kn,
                      const __bf16* __restrict__ vt,   // [H][HD][N] transposed
                      const int* __restrict__ qb, const int* __restrict__ kb,
                      const unsigned* __restrict__ anyn,
                      __bf16* __restrict__ outp, int N) {
  __shared__ int skb[2048];
  __shared__ __align__(16) __bf16 sp[8][16 * 32];

  int h   = blockIdx.y;
  int l   = threadIdx.x & 31;
  int wid = threadIdx.x >> 5;
  int q0  = blockIdx.x * 128 + wid * 16;
  for (int i = threadIdx.x; i < N; i += 256) skb[i] = kb[(size_t)h * N + i];
  __syncthreads();

  int lane16 = l & 15, half = l >> 4;
  const __bf16* qrow = qn + ((size_t)h * N + q0 + lane16) * HD + half * 16;
  bf16x16 qa0 = ld_afrag(qrow);        // d = 0..31
  bf16x16 qa1 = ld_afrag(qrow + 32);   // d = 32..63

  float rmax[8], rsum[8]; int cnt[8], qbr[8]; unsigned hn[8];
#pragma unroll
  for (int r = 0; r < 8; ++r) {
    rmax[r] = -INFINITY; rsum[r] = 0.f; cnt[r] = 0;
    int m  = q0 + (half << 3) + r;
    qbr[r] = qb[(size_t)h * N + m];
    hn[r]  = anyn[(size_t)h * N + m];
  }
  f32x8 o0 = zero8(), o1 = zero8(), o2 = zero8(), o3 = zero8();

  const __bf16* knh = kn + (size_t)h * N * HD;
  const __bf16* vth = vt + (size_t)h * HD * N;
  __bf16* spw = &sp[wid][0];

  for (int kb0 = 0; kb0 < N; kb0 += 32) {
    // ---- S = qn @ kn^T for a 16x32 block (two 16x16 WMMA column tiles, K=64)
    const __bf16* kp0 = knh + (size_t)(kb0 + lane16) * HD + half * 16;
    const __bf16* kp1 = knh + (size_t)(kb0 + 16 + lane16) * HD + half * 16;
    __builtin_prefetch(kp0 + 32 * HD, 0, 0);     // next k-block of K
    f32x8 s0 = zero8(), s1 = zero8();
    s0 = WMMA_BF16(qa0, ld_contig16(kp0), s0);
    s0 = WMMA_BF16(qa1, ld_contig16(kp0 + 32), s0);
    s1 = WMMA_BF16(qa0, ld_contig16(kp1), s1);
    s1 = WMMA_BF16(qa1, ld_contig16(kp1 + 32), s1);

    int kv0 = skb[kb0 + lane16];
    int kv1 = skb[kb0 + 16 + lane16];

    // ---- ring mask + sequential rank truncation + online softmax (per row)
#pragma unroll
    for (int r = 0; r < 8; ++r) {
      int qv  = qbr[r];
      int dd0 = (qv - kv0) & (NBINS - 1);
      int dd1 = (qv - kv1) & (NBINS - 1);
      int n0  = (dd0 <= 1 || dd0 >= NBINS - 1) ? 1 : 0;
      int n1  = (dd1 <= 1 || dd1 >= NBINS - 1) ? 1 : 0;
      // inclusive prefix sums across the 16-lane segment (key order)
      int p0 = n0, p1 = n1;
#pragma unroll
      for (int d = 1; d < 16; d <<= 1) {
        int t0 = __shfl_up(p0, d, 16);
        int t1 = __shfl_up(p1, d, 16);
        if ((l & 15) >= d) { p0 += t0; p1 += t1; }
      }
      int tot0 = __shfl(p0, 15, 16);
      int tot1 = __shfl(p1, 15, 16);
      int rank0 = cnt[r] + p0;
      int rank1 = cnt[r] + tot0 + p1;
      cnt[r] += tot0 + tot1;
      bool sel0, sel1;
      if (hn[r]) {
        sel0 = n0 && (rank0 <= MAXBIN);
        sel1 = n1 && (rank1 <= MAXBIN);
      } else {                                     // fallback: first MAXBIN keys
        sel0 = (kb0 + lane16) < MAXBIN;
        sel1 = (kb0 + 16 + lane16) < MAXBIN;
      }
      float sc0 = sel0 ? s0[r] * RSCALE : -1e30f;
      float sc1 = sel1 ? s1[r] * RSCALE : -1e30f;

      float bm = fmaxf(sc0, sc1);
#pragma unroll
      for (int off = 8; off; off >>= 1) bm = fmaxf(bm, __shfl_xor(bm, off, 16));
      float nm  = fmaxf(rmax[r], bm);
      float fac = __expf(rmax[r] - nm);
      rmax[r] = nm;
      float e0 = __expf(sc0 - nm), e1 = __expf(sc1 - nm);
      float ps = e0 + e1;
#pragma unroll
      for (int off = 8; off; off >>= 1) ps += __shfl_xor(ps, off, 16);
      rsum[r] = rsum[r] * fac + ps;
      o0[r] *= fac; o1[r] *= fac; o2[r] *= fac; o3[r] *= fac;

      int m = (half << 3) + r;                     // C-layout -> LDS [m][k]
      spw[m * 32 + lane16]      = (__bf16)e0;
      spw[m * 32 + 16 + lane16] = (__bf16)e1;
    }
    asm volatile("s_wait_dscnt 0" ::: "memory");   // P staged in LDS, wave-local

    // ---- reload P in A-fragment layout, accumulate O += P @ V (V^T: contiguous keys)
    bf16x16 pa;
    {
      const __bf16* pp = spw + lane16 * 32 + half * 16;
      union { u32x4 q[2]; bf16x16 v; } u;
      u.q[0] = *(const u32x4*)(pp);
      u.q[1] = *(const u32x4*)(pp + 8);
      pa = u.v;
    }
    const __bf16* vp = vth + (size_t)lane16 * N + kb0 + half * 16;
    __builtin_prefetch(vp + 32, 0, 0);             // next k-block of V^T
    o0 = WMMA_BF16(pa, ld_contig16(vp), o0);
    o1 = WMMA_BF16(pa, ld_contig16(vp + (size_t)16 * N), o1);
    o2 = WMMA_BF16(pa, ld_contig16(vp + (size_t)32 * N), o2);
    o3 = WMMA_BF16(pa, ld_contig16(vp + (size_t)48 * N), o3);
  }

  // ---- epilogue: O /= rowsum, write head-merged bf16 [N][EMBED]
#pragma unroll
  for (int r = 0; r < 8; ++r) {
    float inv = 1.0f / rsum[r];
    int row = q0 + (half << 3) + r;
    size_t base = (size_t)row * EMBED + h * HD + lane16;
    outp[base + 0]  = (__bf16)(o0[r] * inv);
    outp[base + 16] = (__bf16)(o1[r] * inv);
    outp[base + 32] = (__bf16)(o2[r] * inv);
    outp[base + 48] = (__bf16)(o3[r] * inv);
  }
}

// ---------------------------------------------------------------- host orchestration
extern "C" void kernel_launch(void* const* d_in, const int* in_sizes, int n_in,
                              void* d_out, int out_size, void* d_ws, size_t ws_size,
                              hipStream_t stream) {
  const float* query = (const float*)d_in[0];
  const float* key   = (const float*)d_in[1];
  const float* value = (const float*)d_in[2];
  const float* Wq    = (const float*)d_in[3];
  const float* bq    = (const float*)d_in[4];
  const float* Wk    = (const float*)d_in[5];
  const float* bk    = (const float*)d_in[6];
  const float* Wv    = (const float*)d_in[7];
  const float* bv    = (const float*)d_in[8];
  const float* Wo    = (const float*)d_in[9];
  const float* bo    = (const float*)d_in[10];
  const float* cen   = (const float*)d_in[11];

  const int N = in_sizes[0] / EMBED;      // 2048 tokens (B=1)

  char* ws = (char*)d_ws;
  size_t off = 0;
  auto alloc = [&](size_t bytes) -> char* {
    char* p = ws + off;
    off += (bytes + 255) & ~(size_t)255;
    return p;
  };
  __bf16* xq   = (__bf16*)alloc((size_t)N * EMBED * 2);
  __bf16* xk   = (__bf16*)alloc((size_t)N * EMBED * 2);
  __bf16* xv   = (__bf16*)alloc((size_t)N * EMBED * 2);
  __bf16* wq   = (__bf16*)alloc((size_t)EMBED * EMBED * 2);
  __bf16* wk   = (__bf16*)alloc((size_t)EMBED * EMBED * 2);
  __bf16* wv   = (__bf16*)alloc((size_t)EMBED * EMBED * 2);
  __bf16* wo   = (__bf16*)alloc((size_t)EMBED * EMBED * 2);
  __bf16* cenb = (__bf16*)alloc((size_t)NBINS * HD * 2);
  float*  Qf   = (float*)alloc((size_t)NHEADS * N * HD * 4);
  float*  Kf   = (float*)alloc((size_t)NHEADS * N * HD * 4);
  float*  Vf   = (float*)alloc((size_t)NHEADS * N * HD * 4);
  __bf16* qnb  = (__bf16*)alloc((size_t)NHEADS * N * HD * 2);
  __bf16* knb  = (__bf16*)alloc((size_t)NHEADS * N * HD * 2);
  __bf16* vtb  = (__bf16*)alloc((size_t)NHEADS * HD * N * 2);
  int*    qbi  = (int*)alloc((size_t)NHEADS * N * 4);
  int*    kbi  = (int*)alloc((size_t)NHEADS * N * 4);
  unsigned* any = (unsigned*)alloc((size_t)NHEADS * N * 4);
  __bf16* att  = (__bf16*)alloc((size_t)N * EMBED * 2);

  const int nx = N * EMBED, nw = EMBED * EMBED;
  cvt_f32_bf16<<<1024, 256, 0, stream>>>(query, xq, nx);
  cvt_f32_bf16<<<1024, 256, 0, stream>>>(key,   xk, nx);
  cvt_f32_bf16<<<1024, 256, 0, stream>>>(value, xv, nx);
  cvt_f32_bf16<<<512,  256, 0, stream>>>(Wq, wq, nw);
  cvt_f32_bf16<<<512,  256, 0, stream>>>(Wk, wk, nw);
  cvt_f32_bf16<<<512,  256, 0, stream>>>(Wv, wv, nw);
  cvt_f32_bf16<<<512,  256, 0, stream>>>(Wo, wo, nw);
  cvt_f32_bf16<<<128,  256, 0, stream>>>(cen, cenb, NBINS * HD);

  const int gemm_blocks = (EMBED / 16) * (N / 16) / 8;
  gemm_wmma_bf16<<<gemm_blocks, 256, 0, stream>>>(xq, wq, bq, Qf, N, EMBED, EMBED, 1);
  gemm_wmma_bf16<<<gemm_blocks, 256, 0, stream>>>(xk, wk, bk, Kf, N, EMBED, EMBED, 1);
  gemm_wmma_bf16<<<gemm_blocks, 256, 0, stream>>>(xv, wv, bv, Vf, N, EMBED, EMBED, 1);

  const int rows = NHEADS * N;
  norm_kernel<<<rows / 8, 256, 0, stream>>>(Qf, qnb);
  norm_kernel<<<rows / 8, 256, 0, stream>>>(Kf, knb);
  bin_wmma_kernel<<<rows / 128, 256, 0, stream>>>(qnb, cenb, qbi);
  bin_wmma_kernel<<<rows / 128, 256, 0, stream>>>(knb, cenb, kbi);

  dim3 tgrid(N / 16, NHEADS);
  transpose_v_kernel<<<tgrid, 256, 0, stream>>>(Vf, vtb, N);

  neigh_any_kernel<<<NHEADS, 256, 0, stream>>>(qbi, kbi, any, N);

  dim3 agrid(N / 128, NHEADS);
  attn_wmma_kernel<<<agrid, 256, 0, stream>>>(qnb, knb, vtb, qbi, kbi, any, att, N);

  gemm_wmma_bf16<<<gemm_blocks, 256, 0, stream>>>(att, wo, bo, (float*)d_out, N, EMBED, EMBED, 0);
}
